// MultiHeadAttention_46110768889919
// MI455X (gfx1250) — matrix operand order repaired and checked by reading, not compiled, not run on hardware
//
#include <hip/hip_runtime.h>
#include <hip/hip_bf16.h>

#define DEV __device__ __forceinline__

constexpr int B  = 2;
constexpr int S  = 2048;
constexpr int DM = 1024;   // d_model
constexpr int H  = 16;
constexpr int DH = 64;     // d_head

typedef __attribute__((ext_vector_type(16))) __bf16 v16bf;
typedef __attribute__((ext_vector_type(8)))  __bf16 v8bf;
typedef __attribute__((ext_vector_type(8)))  float  v8f;
typedef __attribute__((ext_vector_type(4)))  unsigned int v4u;
typedef __attribute__((ext_vector_type(8)))  int    v8i;
typedef __attribute__((ext_vector_type(4)))  int    v4i;

#if defined(__has_builtin)
#if __has_builtin(__builtin_amdgcn_tensor_load_to_lds) && __has_builtin(__builtin_amdgcn_s_wait_tensorcnt)
#define HAVE_TDM 1
#endif
#endif
#ifndef HAVE_TDM
#define HAVE_TDM 0
#endif

DEV v8f wmma_bf16(v16bf a, v16bf b, v8f c) {
  return __builtin_amdgcn_wmma_f32_16x16x32_bf16(false, a, false, b, (short)0, c, false, false);
}

// ---- A fragment: 16x32 (MxK), (m,k) at base[m*ld + k] (bf16) ----
DEV v16bf load_a_bf16(const __bf16* base, int ld) {
  const int lane = threadIdx.x & 31;
  const int m = lane & 15, hf = lane >> 4;
  const __bf16* row = base + (size_t)m * ld;
  v16bf a;
#pragma unroll
  for (int i = 0; i < 16; ++i) a[i] = row[hf * 16 + i];
  return a;
}

// ---- B fragment 32x16 (KxN) from K-contiguous storage: (k,n) at base[n*ldn + k] ----
DEV v16bf load_bT_bf16(const __bf16* base, int ldn) {
  const int lane = threadIdx.x & 31;
  const int n = lane & 15, hf = lane >> 4;
  const __bf16* p = base + (size_t)n * ldn + hf * 16;
  v16bf b;
#pragma unroll
  for (int i = 0; i < 16; ++i) b[i] = p[i];
  return b;
}

#if HAVE_TDM
// 2D TDM load: tile == tensor region [rows x width], row stride `stride` (elems),
// element size 2^dsz bytes, into LDS byte offset lds_off. D# per ISA 08 §8.
DEV void tdm_load_2d(unsigned lds_off, const void* gptr, unsigned dsz,
                     unsigned width, unsigned rows, unsigned stride) {
  unsigned long long ga = (unsigned long long)(uintptr_t)gptr;
  v4u g0 = { 1u,                                   // count=1, user desc
             lds_off,
             (unsigned)ga,
             (unsigned)((ga >> 32) & 0x01FFFFFFu) | 0x80000000u };  // addr[56:32] | type=2
  v8i g1 = { (int)(dsz << 16),                     // data_size
             (int)(width << 16),                   // tensor_dim0[15:0]  (bits 79:48)
             (int)(rows  << 16),                   // tensor_dim1[15:0]  (bits 111:80)
             (int)(width << 16),                   // tile_dim0          (bits 127:112)
             (int)rows,                            // tile_dim1          (bits 143:128)
             (int)stride,                          // tensor_dim0_stride[31:0]
             0, 0 };
  v4i gz = { 0, 0, 0, 0 };
#if __clang_major__ >= 23
  v8i gz8 = { 0, 0, 0, 0, 0, 0, 0, 0 };
  __builtin_amdgcn_tensor_load_to_lds(g0, g1, gz, gz, gz8, 0);
#else
  __builtin_amdgcn_tensor_load_to_lds(g0, g1, gz, gz, 0);
#endif
}
#endif

// ============================================================================
// Preprocess A: flat f32 -> bf16, 8 elems/thread (2x float4 in, 16B out)
// ============================================================================
__global__ __launch_bounds__(256) void cvt_bf16_kernel(
    const float* __restrict__ in, __bf16* __restrict__ out) {
  const int i = blockIdx.x * blockDim.x + threadIdx.x;
  const float4* p = (const float4*)in + (size_t)i * 2;
  float4 a = p[0], b = p[1];
  v8bf o;
  o[0] = (__bf16)a.x; o[1] = (__bf16)a.y; o[2] = (__bf16)a.z; o[3] = (__bf16)a.w;
  o[4] = (__bf16)b.x; o[5] = (__bf16)b.y; o[6] = (__bf16)b.z; o[7] = (__bf16)b.w;
  ((v8bf*)out)[i] = o;
}

// ============================================================================
// Preprocess B: transpose+convert. in: M matrices [R x C] f32 row-major,
// out[m][c][r] bf16 (K-contiguous for load_bT). 8 r-values per thread.
// ============================================================================
__global__ __launch_bounds__(256) void cvtT_bf16_kernel(
    const float* __restrict__ in, __bf16* __restrict__ out, int R, int C) {
  const int i = blockIdx.x * blockDim.x + threadIdx.x;
  const int rchunks = R / 8;
  const int r0 = (i % rchunks) * 8;
  const int c  = (i / rchunks) % C;
  const int m  = i / (rchunks * C);
  const float* src = in + ((size_t)m * R + r0) * C + c;
  v8bf o;
#pragma unroll
  for (int j = 0; j < 8; ++j) o[j] = (__bf16)src[(size_t)j * C];
  *(v8bf*)(out + ((size_t)m * C + c) * R + r0) = o;
}

// ============================================================================
// Kernel 1: QKV projection. grid (S/64, B*H, 3), block 128 (4 waves).
// Each wave computes a 16(row) x 64(col) strip: one A fragment feeds 4 WMMAs.
// Q stored scaled; V stored TRANSPOSED [B,H,DH,S] (per-lane contiguous store).
// ============================================================================
__global__ __launch_bounds__(128) void qkv_proj_kernel(
    const __bf16* __restrict__ Xq, const __bf16* __restrict__ Xk, const __bf16* __restrict__ Xv,
    const __bf16* __restrict__ Wqt, const __bf16* __restrict__ Wkt, const __bf16* __restrict__ Wvt,
    const float* __restrict__ bq, const float* __restrict__ bk, const float* __restrict__ bv,
    __bf16* __restrict__ Qb, __bf16* __restrict__ Kb, __bf16* __restrict__ Vt)
{
  const int wave = threadIdx.x >> 5;
  const int s0 = blockIdx.x * 64 + wave * 16;
  const int bh = blockIdx.y;
  const int b  = bh / H, h = bh % H;
  const int which = blockIdx.z;
  const __bf16* x  = (which == 0) ? Xq : (which == 1) ? Xk : Xv;
  const __bf16* Wt = (which == 0) ? Wqt : (which == 1) ? Wkt : Wvt;
  const float* bias = (which == 0) ? bq : (which == 1) ? bk : bv;

  const __bf16* xbase = x + ((size_t)b * S + s0) * DM;
  const __bf16* wbase = Wt + (size_t)h * DH * DM;   // [DH][DM], row = out col

  v8f zero = {};
  v8f c[4] = {zero, zero, zero, zero};
  for (int kk = 0; kk < DM; kk += 32) {
    v16bf a = load_a_bf16(xbase + kk, DM);
#pragma unroll
    for (int nb = 0; nb < 4; ++nb) {
      v16bf bb = load_bT_bf16(wbase + (size_t)(nb * 16) * DM + kk, DM);
      c[nb] = wmma_bf16(a, bb, c[nb]);
    }
  }
  const int lane = threadIdx.x & 31;
  const int n = lane & 15, hf = lane >> 4;

  if (which == 2) {
    // Vt[bh][d = nb*16+n][s0 + m], m = r + 8*hf -> 8 consecutive bf16 per lane
#pragma unroll
    for (int nb = 0; nb < 4; ++nb) {
      const float bias_n = bias[h * DH + nb * 16 + n];
      __bf16* obase = Vt + ((size_t)bh * DH + nb * 16 + n) * S + s0 + 8 * hf;
#pragma unroll
      for (int r = 0; r < 8; ++r) obase[r] = (__bf16)(c[nb][r] + bias_n);
    }
  } else {
    const float sc = (which == 0) ? 0.125f : 1.0f;  // fold 1/sqrt(DH) into Q
    __bf16* out = (which == 0) ? Qb : Kb;
#pragma unroll
    for (int nb = 0; nb < 4; ++nb) {
      const float bias_n = bias[h * DH + nb * 16 + n];
      __bf16* obase = out + ((size_t)bh * S + s0) * DH + nb * 16;
#pragma unroll
      for (int r = 0; r < 8; ++r)
        obase[(size_t)(r + 8 * hf) * DH + n] = (__bf16)((c[nb][r] + bias_n) * sc);
    }
  }
}

// ============================================================================
// Kernel 2: flash attention. grid (S/64, B*H), block 128 (4 waves).
// K/V staged to LDS via TDM (double-buffered) + mask via TDM; fallback =
// cooperative vector loads. Online softmax with wave32 shfl_xor reductions.
// LDS layout (dynamic): sK 2x4KB @0, sVt 2x4KB @8192, sM 8KB @16384, sP 4KB @24576
// ============================================================================
constexpr unsigned SMEM_BYTES = 28672;

__global__ __launch_bounds__(128) void flash_attn_kernel(
    const __bf16* __restrict__ Qb, const __bf16* __restrict__ Kb,
    const __bf16* __restrict__ Vt, const float* __restrict__ mask,
    __bf16* __restrict__ Ob)
{
  extern __shared__ char smem[];
  __bf16* sK  = (__bf16*)(smem);           // [2][32][64]  (kv row, d)
  __bf16* sVt = (__bf16*)(smem + 8192);    // [2][64][32]  (d, kv row)
  float*  sM  = (float*)(smem + 16384);    // [64][32]
  __bf16* sP  = (__bf16*)(smem + 24576);   // [4][16][32]

  const int q0   = blockIdx.x * 64;
  const int bh   = blockIdx.y;
  const int wave = threadIdx.x >> 5;
  const int lane = threadIdx.x & 31;
  const int n = lane & 15, hf = lane >> 4;
  const int qrow0 = q0 + wave * 16;
  const int t = threadIdx.x;

  const __bf16* Qbase  = Qb + ((size_t)bh * S + qrow0) * DH;
  const __bf16* Kbase  = Kb + (size_t)bh * S * DH;
  const __bf16* Vtbase = Vt + (size_t)bh * DH * S;

  v16bf aQ0 = load_a_bf16(Qbase + 0,  DH);
  v16bf aQ1 = load_a_bf16(Qbase + 32, DH);

  v8f zero = {};
  v8f acc[4] = {zero, zero, zero, zero};
  float mrun[8], lrun[8];
#pragma unroll
  for (int r = 0; r < 8; ++r) { mrun[r] = -1e30f; lrun[r] = 0.0f; }

  constexpr int NIT = S / 32;

#if HAVE_TDM
  if (wave == 0) {
    tdm_load_2d(0u,    Kbase,  1u, 64u, 32u, 64u);          // K tile 0 -> buf0
    tdm_load_2d(8192u, Vtbase, 1u, 32u, 64u, (unsigned)S);  // Vt tile 0 -> buf0
  }
#endif

  for (int it = 0; it < NIT; ++it) {
    const int kv0 = it * 32;
    const int cur = it & 1;
    __bf16* cK  = sK  + cur * 2048;
    __bf16* cVt = sVt + cur * 2048;

#if HAVE_TDM
    if (wave == 0) {
      tdm_load_2d(16384u, mask + (size_t)q0 * S + kv0, 2u, 32u, 64u, (unsigned)S);
      if (it + 1 < NIT) {
        const int nxt = cur ^ 1;
        tdm_load_2d((unsigned)(nxt * 4096),        Kbase  + (size_t)(kv0 + 32) * DH, 1u, 64u, 32u, 64u);
        tdm_load_2d((unsigned)(8192 + nxt * 4096), Vtbase + (kv0 + 32),              1u, 32u, 64u, (unsigned)S);
        __builtin_amdgcn_s_wait_tensorcnt((short)2);   // current K/V + mask complete
      } else {
        __builtin_amdgcn_s_wait_tensorcnt((short)0);
      }
    }
#else
    {
      // K: contiguous 4KB
      const uint4* srcK = (const uint4*)(Kbase + (size_t)kv0 * DH);
      uint4* dK = (uint4*)cK;
      dK[t * 2] = srcK[t * 2];
      dK[t * 2 + 1] = srcK[t * 2 + 1];
      // Vt: 64 rows x 32 cols, row stride S
      const int vr = t >> 1, vc = (t & 1) * 16;
      const uint4* srcV = (const uint4*)(Vtbase + (size_t)vr * S + kv0 + vc);
      uint4* dV = (uint4*)(cVt + vr * 32 + vc);
      dV[0] = srcV[0];
      dV[1] = srcV[1];
      // mask tile 64x32 f32
      const int mr = t >> 1, mc = (t & 1) * 16;
      const float4* srcM = (const float4*)(mask + (size_t)(q0 + mr) * S + kv0 + mc);
      float4* dM = (float4*)(sM + mr * 32 + mc);
#pragma unroll
      for (int i = 0; i < 4; ++i) dM[i] = srcM[i];
    }
#endif
    __syncthreads();

    // -------- scores: 16(q) x 32(kv) --------
    v8f s0 = {}, s1 = {};
    s0 = wmma_bf16(aQ0, load_bT_bf16(cK + 0 * 64 + 0,  64), s0);
    s0 = wmma_bf16(aQ1, load_bT_bf16(cK + 0 * 64 + 32, 64), s0);
    s1 = wmma_bf16(aQ0, load_bT_bf16(cK + 16 * 64 + 0,  64), s1);
    s1 = wmma_bf16(aQ1, load_bT_bf16(cK + 16 * 64 + 32, 64), s1);

    // -------- multiplicative mask --------
    const int rl = wave * 16 + 8 * hf;
#pragma unroll
    for (int r = 0; r < 8; ++r) {
      s0[r] *= sM[(rl + r) * 32 + n];
      s1[r] *= sM[(rl + r) * 32 + 16 + n];
    }

    // -------- online softmax --------
    float mnew[8], alpha[8];
#pragma unroll
    for (int r = 0; r < 8; ++r) {
      float v = fmaxf(s0[r], s1[r]);
#pragma unroll
      for (int off = 1; off < 16; off <<= 1)
        v = fmaxf(v, __shfl_xor(v, off, 32));
      mnew[r]  = fmaxf(mrun[r], v);
      alpha[r] = __expf(mrun[r] - mnew[r]);
      mrun[r]  = mnew[r];
    }
#pragma unroll
    for (int r = 0; r < 8; ++r) {
      float p0 = __expf(s0[r] - mnew[r]);
      float p1 = __expf(s1[r] - mnew[r]);
      s0[r] = p0; s1[r] = p1;
      float rs = p0 + p1;
#pragma unroll
      for (int off = 1; off < 16; off <<= 1)
        rs += __shfl_xor(rs, off, 32);
      lrun[r] = lrun[r] * alpha[r] + rs;
    }
#pragma unroll
    for (int nb2 = 0; nb2 < 4; ++nb2)
#pragma unroll
      for (int r = 0; r < 8; ++r) acc[nb2][r] *= alpha[r];

    // -------- P (C layout) -> LDS -> A fragment; O += P · V --------
    __bf16* wP = sP + wave * 512;
#pragma unroll
    for (int r = 0; r < 8; ++r) {
      wP[(8 * hf + r) * 32 + n]      = (__bf16)s0[r];
      wP[(8 * hf + r) * 32 + 16 + n] = (__bf16)s1[r];
    }
    v16bf aP = load_a_bf16(wP, 32);
#pragma unroll
    for (int nb2 = 0; nb2 < 4; ++nb2) {
      v16bf bV = load_bT_bf16(cVt + (nb2 * 16) * 32, 32);  // (k=kv, n=d) contiguous
      acc[nb2] = wmma_bf16(aP, bV, acc[nb2]);
    }
    __syncthreads();
  }

  // -------- normalize, store concat layout [B, S, H*DH] --------
  const int b = bh / H, h = bh % H;
  __bf16* obase = Ob + ((size_t)b * S + qrow0) * DM + h * DH;
#pragma unroll
  for (int nb2 = 0; nb2 < 4; ++nb2) {
#pragma unroll
    for (int r = 0; r < 8; ++r) {
      float o = acc[nb2][r] / lrun[r];
      obase[(size_t)(8 * hf + r) * DM + nb2 * 16 + n] = (__bf16)o;
    }
  }
}

// ============================================================================
// Kernel 3: output projection. grid ((B*S)/64, DM/64), block 128 (4 waves).
// Each wave computes a 16(row) x 64(col) strip: one A fragment feeds 4 WMMAs.
// ============================================================================
__global__ __launch_bounds__(128) void out_proj_kernel(
    const __bf16* __restrict__ Ob, const __bf16* __restrict__ Wot,
    const float* __restrict__ bo, float* __restrict__ out)
{
  const int wave = threadIdx.x >> 5;
  const int row0 = blockIdx.x * 64 + wave * 16;
  const int col0 = blockIdx.y * 64;
  const __bf16* abase = Ob + (size_t)row0 * DM;
  const __bf16* wbase = Wot + (size_t)col0 * DM;   // row = out col (pre-transposed)

  v8f zero = {};
  v8f c[4] = {zero, zero, zero, zero};
  for (int kk = 0; kk < DM; kk += 32) {
    v16bf a = load_a_bf16(abase + kk, DM);
#pragma unroll
    for (int nb = 0; nb < 4; ++nb) {
      v16bf bb = load_bT_bf16(wbase + (size_t)(nb * 16) * DM + kk, DM);
      c[nb] = wmma_bf16(a, bb, c[nb]);
    }
  }
  const int lane = threadIdx.x & 31;
  const int n = lane & 15, hf = lane >> 4;
#pragma unroll
  for (int nb = 0; nb < 4; ++nb) {
    const float bn = bo[col0 + nb * 16 + n];
    float* obase = out + (size_t)row0 * DM + col0 + nb * 16;
#pragma unroll
    for (int r = 0; r < 8; ++r)
      obase[(size_t)(8 * hf + r) * DM + n] = c[nb][r] + bn;
  }
}

// ============================================================================
extern "C" void kernel_launch(void* const* d_in, const int* in_sizes, int n_in,
                              void* d_out, int out_size, void* d_ws, size_t ws_size,
                              hipStream_t stream) {
  const float* q    = (const float*)d_in[0];
  const float* k    = (const float*)d_in[1];
  const float* v    = (const float*)d_in[2];
  const float* mask = (const float*)d_in[3];
  const float* Wq   = (const float*)d_in[4];
  const float* bq   = (const float*)d_in[5];
  const float* Wk   = (const float*)d_in[6];
  const float* bk   = (const float*)d_in[7];
  const float* Wv   = (const float*)d_in[8];
  const float* bv   = (const float*)d_in[9];
  const float* Wo   = (const float*)d_in[10];
  const float* bo   = (const float*)d_in[11];
  float* out = (float*)d_out;

  const size_t nX   = (size_t)B * S * DM;      // 4 Mi
  const size_t nW   = (size_t)H * DM * DH;     // 1 Mi
  const size_t nQKV = (size_t)B * H * S * DH;  // 4 Mi
  char* ws = (char*)d_ws;
  __bf16* Xq  = (__bf16*)(ws);
  __bf16* Xk  = (__bf16*)(ws + 2 * nX);
  __bf16* Xv  = (__bf16*)(ws + 4 * nX);
  __bf16* Wqt = (__bf16*)(ws + 6 * nX);
  __bf16* Wkt = (__bf16*)(ws + 6 * nX + 2 * nW);
  __bf16* Wvt = (__bf16*)(ws + 6 * nX + 4 * nW);
  __bf16* Wot = (__bf16*)(ws + 6 * nX + 6 * nW);
  __bf16* Qb  = (__bf16*)(ws + 6 * nX + 8 * nW);
  __bf16* Kb  = (__bf16*)(ws + 6 * nX + 8 * nW + 2 * nQKV);
  __bf16* Vt  = (__bf16*)(ws + 6 * nX + 8 * nW + 4 * nQKV);
  __bf16* Ob  = (__bf16*)(ws + 6 * nX + 8 * nW + 6 * nQKV);

  // ---- preprocess: convert activations, transpose+convert weights ----
  const int nb8 = (int)(nX / 8);  // 524288 -> 2048 blocks
  cvt_bf16_kernel<<<nb8 / 256, 256, 0, stream>>>(q, Xq);
  cvt_bf16_kernel<<<nb8 / 256, 256, 0, stream>>>(k, Xk);
  cvt_bf16_kernel<<<nb8 / 256, 256, 0, stream>>>(v, Xv);
  const int nwT = (int)(nW / 8);  // 131072 -> 512 blocks
  cvtT_bf16_kernel<<<nwT / 256, 256, 0, stream>>>(Wq, Wqt, DM, DH);
  cvtT_bf16_kernel<<<nwT / 256, 256, 0, stream>>>(Wk, Wkt, DM, DH);
  cvtT_bf16_kernel<<<nwT / 256, 256, 0, stream>>>(Wv, Wvt, DM, DH);
  const int noT = (int)((size_t)DM * DM / 8);
  cvtT_bf16_kernel<<<noT / 256, 256, 0, stream>>>(Wo, Wot, DM, DM);

  dim3 g1(S / 64, B * H, 3);
  qkv_proj_kernel<<<g1, 128, 0, stream>>>(Xq, Xk, Xv, Wqt, Wkt, Wvt, bq, bk, bv, Qb, Kb, Vt);

  dim3 g2(S / 64, B * H);
  flash_attn_kernel<<<g2, 128, SMEM_BYTES, stream>>>(Qb, Kb, Vt, mask, Ob);

  dim3 g3((B * S) / 64, DM / 64);
  out_proj_kernel<<<g3, 128, 0, stream>>>(Ob, Wot, bo, out);
}
